// CRLI_16492674417367
// MI455X (gfx1250) — matrix operand order repaired and checked by reading, not compiled, not run on hardware
//
#include <hip/hip_runtime.h>
#include <hip/hip_bf16.h>
#include <math.h>

typedef __attribute__((ext_vector_type(16))) _Float16 v16h;
typedef __attribute__((ext_vector_type(8)))  float    v8f;

#define BATCH 32768
#define SEQ   16
#define HID   16
#define BS    (BATCH * SEQ)

// ---- WMMA helper: D = A(16x32 f16) x B(32x16 f16) + C(16x16 f32) ----
__device__ __forceinline__ v8f wmma16(v16h a, v16h b, v8f c) {
  return __builtin_amdgcn_wmma_f32_16x16x32_f16(false, a, false, b, (short)0, c,
                                                false, false);
}

// ---- branch-free fast transcendentals on CDNA5 TRANS hardware ----
__device__ __forceinline__ float fexp2(float x) {
#if __has_builtin(__builtin_amdgcn_exp2f)
  return __builtin_amdgcn_exp2f(x);   // v_exp_f32
#else
  return exp2f(x);
#endif
}
__device__ __forceinline__ float frcp(float x) {
#if __has_builtin(__builtin_amdgcn_rcpf)
  return __builtin_amdgcn_rcpf(x);    // v_rcp_f32
#else
  return 1.f / x;
#endif
}
__device__ __forceinline__ float sigm(float x) {
  // 1/(1+2^(-x*log2e)); saturates correctly at +/-inf, no branches
  return frcp(1.f + fexp2(-1.44269504088896340736f * x));
}
__device__ __forceinline__ float ftanh(float x) {
#if __has_builtin(__builtin_amdgcn_tanhf)
  return __builtin_amdgcn_tanhf(x);   // v_tanh_f32 (CDNA5 TRANS op)
#else
  float xc = fminf(fmaxf(x, -10.f), 10.f);
  float t = fexp2(2.88539008177792681472f * xc);   // e^(2x)
  return (t - 1.f) * frcp(t + 1.f);
#endif
}

// A fragment builder (16x32 A, this file's contiguous k order):
//   lanes 0-15: m=lane, elements hold K=0..15
//   lanes 16-31: m=lane-16, elements hold K=16..31
// Tile: rows [0,nrows) of W (ldw stride), cols [0,ncols); optional extra
// column ncols = e0[row] and ncols+1 = e1[row]; everything else zero.
__device__ __forceinline__ v16h make_a(const float* W, int ldw, int nrows, int ncols,
                                       const float* e0, const float* e1, int lane) {
  const int m = lane & 15, half = lane >> 4;
  v16h a;
  #pragma unroll
  for (int e = 0; e < 16; ++e) {
    const int k = 16 * half + e;
    float v = 0.f;
    if (m < nrows) {
      if (k < ncols)                 v = W[m * ldw + k];
      else if (e0 && k == ncols)     v = e0[m];
      else if (e1 && k == ncols + 1) v = e1[m];
    }
    a[e] = (_Float16)v;
  }
  return a;
}

// Exchange the two lane-halves of a canonical state (lane(n,h) holds s[8h+r]).
__device__ __forceinline__ void swap_half(const float s[8], float o[8], int lane) {
  #pragma unroll
  for (int r = 0; r < 8; ++r) o[r] = __shfl(s[r], lane ^ 16, 32);
}

// B fragment (32x16 B, this file's contiguous k order): lanes 0-15 hold K=0..15 (col n=lane),
// lanes 16-31 hold K=16..31. Rows 0-15 = state vector, row16 = x16, row17 = x17.
__device__ __forceinline__ v16h make_b16(const float st[8], const float oth[8],
                                         float x16, float x17, int half) {
  v16h b;
  #pragma unroll
  for (int e = 0; e < 16; ++e) {
    float lo = (e < 8) ? st[e] : oth[e - 8];
    float hi = (e == 0) ? x16 : ((e == 1) ? x17 : 0.f);
    b[e] = (_Float16)(half ? hi : lo);
  }
  return b;
}

__device__ __forceinline__ void lstm_update(const v8f& gi, const v8f& gf, const v8f& gg,
                                            const v8f& go, float h[8], float c[8]) {
  #pragma unroll
  for (int r = 0; r < 8; ++r) {
    float cn = sigm(gf[r]) * c[r] + sigm(gi[r]) * ftanh(gg[r]);
    c[r] = cn;
    h[r] = sigm(go[r]) * ftanh(cn);
  }
}

__device__ __forceinline__ void disc_act(const v8f& gi, const v8f& gg, const v8f& go,
                                         float o[8]) {
  #pragma unroll
  for (int r = 0; r < 8; ++r) {
    float cc = sigm(gi[r]) * ftanh(gg[r]);
    o[r] = sigm(go[r]) * ftanh(cc);
  }
}

__global__ __launch_bounds__(256) void crli_kernel(
    const float* __restrict__ values, const int* __restrict__ masks,
    const float* __restrict__ gfWih, const float* __restrict__ gfWhh,
    const float* __restrict__ gfb, const float* __restrict__ gbWih,
    const float* __restrict__ gbWhh, const float* __restrict__ gbb,
    const float* __restrict__ impW, const float* __restrict__ impb,
    const float* __restrict__ fcW, const float* __restrict__ fcb,
    const float* __restrict__ decWih, const float* __restrict__ decWhh,
    const float* __restrict__ decb, const float* __restrict__ decoW,
    const float* __restrict__ decob, const float* __restrict__ dscoW,
    const float* __restrict__ dscob,
    const float* __restrict__ dW0, const float* __restrict__ db0,
    const float* __restrict__ dW1, const float* __restrict__ db1,
    const float* __restrict__ dW2, const float* __restrict__ db2,
    const float* __restrict__ dW3, const float* __restrict__ db3,
    const float* __restrict__ dW4, const float* __restrict__ db4,
    float* __restrict__ out) {
  const int lane = threadIdx.x & 31;
  const int wid  = threadIdx.x >> 5;
  const int half = lane >> 4;
  const int n    = lane & 15;
  const int b0   = (blockIdx.x * 8 + wid) * 16;

  __shared__ float sv[8][256];  // values tile, [s*16+n]
  __shared__ float sm[8][256];  // mask tile
  __shared__ float si[8][256];  // imputed tile

  v8f zf;
  #pragma unroll
  for (int r = 0; r < 8; ++r) zf[r] = 0.f;
  float zero8[8];
  #pragma unroll
  for (int r = 0; r < 8; ++r) zero8[r] = 0.f;

  // ---- stage value/mask tiles (lane(n,h) owns s=8h..8h+7 of row b0+n) ----
  #pragma unroll
  for (int r = 0; r < 8; ++r) {
    int s = 8 * half + r;
    sv[wid][s * 16 + n] = values[(b0 + n) * SEQ + s];
    sm[wid][s * 16 + n] = (float)masks[(b0 + n) * SEQ + s];
  }
  __syncthreads();

  // =============== Phase 1: bidirectional generator LSTM ===============
  // A chunk c: rows = gates 16c..16c+15; cols0-15 = Whh, col16 = Wih, col17 = b
  v16h Af[4], Ab[4];
  #pragma unroll
  for (int c = 0; c < 4; ++c) {
    Af[c] = make_a(gfWhh + c * 16 * HID, HID, 16, 16, gfWih + 16 * c, gfb + 16 * c, lane);
    Ab[c] = make_a(gbWhh + c * 16 * HID, HID, 16, 16, gbWih + 16 * c, gbb + 16 * c, lane);
  }
  v16h Aimp = make_a(impW, HID, 1, 16, impb, nullptr, lane);

  float hf[8], cf[8], hb[8], cb[8];
  #pragma unroll
  for (int r = 0; r < 8; ++r) { hf[r] = cf[r] = hb[r] = cb[r] = 0.f; }

  {  // init cell: x = +/-128, h = c = 0
    v16h Bf = make_b16(zero8, zero8, 128.f, 1.f, half);
    v16h Bb = make_b16(zero8, zero8, -128.f, 1.f, half);
    v8f i0 = wmma16(Af[0], Bf, zf), f0 = wmma16(Af[1], Bf, zf);
    v8f g0 = wmma16(Af[2], Bf, zf), o0 = wmma16(Af[3], Bf, zf);
    lstm_update(i0, f0, g0, o0, hf, cf);
    v8f i1 = wmma16(Ab[0], Bb, zf), f1 = wmma16(Ab[1], Bb, zf);
    v8f g1 = wmma16(Ab[2], Bb, zf), o1 = wmma16(Ab[3], Bb, zf);
    lstm_update(i1, f1, g1, o1, hb, cb);
  }

  #pragma unroll 1
  for (int t = 0; t < SEQ; ++t) {
    float of[8], ob[8];
    swap_half(hf, of, lane);
    swap_half(hb, ob, lane);
    // imputation value: hv = h . impW + impb  (row 0 of imp tile)
    v16h Bhf = make_b16(hf, of, 1.f, 0.f, half);
    v16h Bhb = make_b16(hb, ob, 1.f, 0.f, half);
    v8f tif = wmma16(Aimp, Bhf, zf);
    v8f tib = wmma16(Aimp, Bhb, zf);
    float xv = sv[wid][t * 16 + n];
    float mv = sm[wid][t * 16 + n];
    float ccf = (1.f - mv) * xv + mv * tif[0];  // valid on half 0
    float ccb = (1.f - mv) * xv + mv * tib[0];
    float ccfA = __shfl(ccf, n, 32);            // broadcast half0 -> all
    float ccbA = __shfl(ccb, n, 32);
    v16h Bf = make_b16(hf, of, ccfA, 1.f, half);  // [h; cc; 1]
    v16h Bb = make_b16(hb, ob, ccbA, 1.f, half);
    v8f fi = wmma16(Af[0], Bf, zf), ff = wmma16(Af[1], Bf, zf);
    v8f fg = wmma16(Af[2], Bf, zf), fo = wmma16(Af[3], Bf, zf);
    lstm_update(fi, ff, fg, fo, hf, cf);
    v8f bi = wmma16(Ab[0], Bb, zf), bf_ = wmma16(Ab[1], Bb, zf);
    v8f bg = wmma16(Ab[2], Bb, zf), bo = wmma16(Ab[3], Bb, zf);
    lstm_update(bi, bf_, bg, bo, hb, cb);
  }

  float hs[8];
  #pragma unroll
  for (int r = 0; r < 8; ++r) hs[r] = hf[r] + hb[r];

  // imputed output (note H==S==16: imputations[b][s] = hsum[s])
  #pragma unroll
  for (int r = 0; r < 8; ++r) {
    int s = 8 * half + r;
    float mv = sm[wid][s * 16 + n];
    float xv = sv[wid][s * 16 + n];
    float iv = hs[r] * (1.f - mv) + xv * mv;
    out[(b0 + n) * SEQ + s] = iv;
    si[wid][s * 16 + n] = iv;
  }

  {  // latent = hsum @ fcW^T + fcb
    float oh[8];
    swap_half(hs, oh, lane);
    v16h Bs = make_b16(hs, oh, 1.f, 0.f, half);
    v16h Afc = make_a(fcW, 16, 16, 16, fcb, nullptr, lane);
    v8f lat = wmma16(Afc, Bs, zf);
    #pragma unroll
    for (int r = 0; r < 8; ++r) out[2 * BS + (b0 + n) * 16 + 8 * half + r] = lat[r];
  }

  // =============== Phase 2: decoder LSTM ===============
  v16h Adh[4], Adx[4];
  #pragma unroll
  for (int c = 0; c < 4; ++c) {
    Adh[c] = make_a(decWhh + c * 16 * 16, 16, 16, 16, nullptr, nullptr, lane);
    Adx[c] = make_a(decWih + c * 16 * 16, 16, 16, 16, decb + 16 * c, nullptr, lane);
  }
  v16h Adeco = make_a(decoW, 16, 1, 16, decob, nullptr, lane);

  float hd[8], cd[8];
  #pragma unroll
  for (int r = 0; r < 8; ++r) { hd[r] = cd[r] = 0.f; }
  {  // init cell: x = 128 in all LATENT dims, h = c = 0
    v16h Bx0;
    #pragma unroll
    for (int e = 0; e < 16; ++e) {
      float hi = (e == 0) ? 1.f : 0.f;
      Bx0[e] = (_Float16)(half ? hi : 128.f);
    }
    v16h Bh0 = make_b16(zero8, zero8, 1.f, 0.f, half);
    v8f gi = wmma16(Adx[0], Bx0, wmma16(Adh[0], Bh0, zf));
    v8f gf = wmma16(Adx[1], Bx0, wmma16(Adh[1], Bh0, zf));
    v8f gg = wmma16(Adx[2], Bx0, wmma16(Adh[2], Bh0, zf));
    v8f go = wmma16(Adx[3], Bx0, wmma16(Adh[3], Bh0, zf));
    lstm_update(gi, gf, gg, go, hd, cd);
  }
  #pragma unroll 1
  for (int s = 0; s < SEQ; ++s) {
    float oc[8], oh[8];
    swap_half(cd, oc, lane);
    swap_half(hd, oh, lane);
    v16h Bx = make_b16(cd, oc, 1.f, 0.f, half);  // x = old cell state, + bias row
    v16h Bh = make_b16(hd, oh, 1.f, 0.f, half);  // Adh col16 = 0 -> harmless
    v8f gi = wmma16(Adx[0], Bx, wmma16(Adh[0], Bh, zf));
    v8f gf = wmma16(Adx[1], Bx, wmma16(Adh[1], Bh, zf));
    v8f gg = wmma16(Adx[2], Bx, wmma16(Adh[2], Bh, zf));
    v8f go = wmma16(Adx[3], Bx, wmma16(Adh[3], Bh, zf));
    lstm_update(gi, gf, gg, go, hd, cd);
    float oh2[8];
    swap_half(hd, oh2, lane);
    v16h Bh2 = make_b16(hd, oh2, 1.f, 0.f, half);
    v8f rec = wmma16(Adeco, Bh2, zf);
    if (!half) out[3 * BS + (b0 + n) * SEQ + s] = rec[0];
  }

  // =============== Phase 3: discriminator (pointwise, 16 groups of 16) ====
  __syncthreads();  // si visible

  const int c32[6] = {0, 1, 4, 5, 6, 7};  // i,i,g,g,o,o chunks for hdim=32
  const int c16[3] = {0, 2, 3};           // i,g,o chunks for hdim=16
  v16h A0[6], A1lo[3], A1hi[3], A2[2], A3[3], A4[6];
  #pragma unroll
  for (int j = 0; j < 6; ++j)
    A0[j] = make_a(dW0 + c32[j] * 16, 1, 16, 1, db0 + c32[j] * 16, nullptr, lane);
  #pragma unroll
  for (int j = 0; j < 3; ++j) {
    int c = c16[j];
    A1lo[j] = make_a(dW1 + c * 16 * 32, 32, 16, 16, db1 + c * 16, nullptr, lane);
    A1hi[j] = make_a(dW1 + c * 16 * 32 + 16, 32, 16, 16, nullptr, nullptr, lane);
  }
  A2[0] = make_a(dW2, 16, 16, 16, db2, nullptr, lane);
  A2[1] = make_a(dW2 + 16 * 16, 16, 16, 16, db2 + 16, nullptr, lane);
  #pragma unroll
  for (int j = 0; j < 3; ++j)
    A3[j] = make_a(dW3 + c16[j] * 16 * 8, 8, 16, 8, db3 + c16[j] * 16, nullptr, lane);
  #pragma unroll
  for (int j = 0; j < 6; ++j)
    A4[j] = make_a(dW4 + c32[j] * 16 * 16, 16, 16, 16, db4 + c32[j] * 16, nullptr, lane);
  v16h Aolo = make_a(dscoW, 32, 1, 16, dscob, nullptr, lane);
  v16h Aohi = make_a(dscoW + 16, 32, 1, 16, nullptr, nullptr, lane);

  #pragma unroll 1
  for (int s = 0; s < SEQ; ++s) {
    float xv = si[wid][s * 16 + n];
    // L0: in=1 -> B rows: K0 = x, K1 = 1
    v16h B0;
    #pragma unroll
    for (int e = 0; e < 16; ++e) {
      float lo = (e == 0) ? xv : ((e == 1) ? 1.f : 0.f);
      B0[e] = (_Float16)(half ? 0.f : lo);
    }
    v8f ti0 = wmma16(A0[0], B0, zf), ti1 = wmma16(A0[1], B0, zf);
    v8f tg0 = wmma16(A0[2], B0, zf), tg1 = wmma16(A0[3], B0, zf);
    v8f to0 = wmma16(A0[4], B0, zf), to1 = wmma16(A0[5], B0, zf);
    float st0[8], st1[8];
    disc_act(ti0, tg0, to0, st0);  // out[0..15]
    disc_act(ti1, tg1, to1, st1);  // out[16..31]

    // L1: in=32 -> lo (x[0:16] + bias row) and hi (x[16:32]) WMMAs
    float o0[8], o1[8];
    swap_half(st0, o0, lane);
    swap_half(st1, o1, lane);
    v16h Blo = make_b16(st0, o0, 1.f, 0.f, half);
    v16h Bhi = make_b16(st1, o1, 0.f, 0.f, half);
    v8f g1i = wmma16(A1lo[0], Blo, wmma16(A1hi[0], Bhi, zf));
    v8f g1g = wmma16(A1lo[1], Blo, wmma16(A1hi[1], Bhi, zf));
    v8f g1o = wmma16(A1lo[2], Blo, wmma16(A1hi[2], Bhi, zf));
    float s16a[8];
    disc_act(g1i, g1g, g1o, s16a);

    // L2: in=16, hdim=8: chunk0 = i|f, chunk1 = g|o (split across halves)
    float o16a[8];
    swap_half(s16a, o16a, lane);
    v16h B2 = make_b16(s16a, o16a, 1.f, 0.f, half);
    v8f t20 = wmma16(A2[0], B2, zf);
    v8f t21 = wmma16(A2[1], B2, zf);
    float s8[8];
    #pragma unroll
    for (int r = 0; r < 8; ++r) {
      float cc = sigm(t20[r]) * ftanh(t21[r]);  // half0: sig(i)*tanh(g)
      float ccp = __shfl(cc, n, 32);            // fetch half0's cc
      float ov = sigm(t21[r]) * ftanh(ccp);     // half1: sig(o)*tanh(cc)
      s8[r] = __shfl(ov, n + 16, 32);           // broadcast half1's result
    }

    // L3: in=8 -> B rows: K0-7 = x, K8 = 1
    v16h B3;
    #pragma unroll
    for (int e = 0; e < 16; ++e) {
      float lo = (e < 8) ? s8[e] : ((e == 8) ? 1.f : 0.f);
      B3[e] = (_Float16)(half ? 0.f : lo);
    }
    v8f g3i = wmma16(A3[0], B3, zf);
    v8f g3g = wmma16(A3[1], B3, zf);
    v8f g3o = wmma16(A3[2], B3, zf);
    float s16b[8];
    disc_act(g3i, g3g, g3o, s16b);

    // L4: in=16, hdim=32
    float o16b[8];
    swap_half(s16b, o16b, lane);
    v16h B4 = make_b16(s16b, o16b, 1.f, 0.f, half);
    v8f fi0 = wmma16(A4[0], B4, zf), fi1 = wmma16(A4[1], B4, zf);
    v8f fg0 = wmma16(A4[2], B4, zf), fg1 = wmma16(A4[3], B4, zf);
    v8f fo0 = wmma16(A4[4], B4, zf), fo1 = wmma16(A4[5], B4, zf);
    float f0[8], f1[8];
    disc_act(fi0, fg0, fo0, f0);
    disc_act(fi1, fg1, fo1, f1);

    // disc_out: in=32 dot -> lo/hi single-row WMMAs
    float q0[8], q1[8];
    swap_half(f0, q0, lane);
    swap_half(f1, q1, lane);
    v16h Bol = make_b16(f0, q0, 1.f, 0.f, half);
    v16h Boh = make_b16(f1, q1, 0.f, 0.f, half);
    v8f dd = wmma16(Aolo, Bol, wmma16(Aohi, Boh, zf));
    if (!half) out[BS + (b0 + n) * SEQ + s] = dd[0];
  }
}

extern "C" void kernel_launch(void* const* d_in, const int* in_sizes, int n_in,
                              void* d_out, int out_size, void* d_ws, size_t ws_size,
                              hipStream_t stream) {
  (void)in_sizes; (void)n_in; (void)out_size; (void)d_ws; (void)ws_size;
  const float* values = (const float*)d_in[0];
  const int*   masks  = (const int*)d_in[1];
  const float* gfWih  = (const float*)d_in[2];
  const float* gfWhh  = (const float*)d_in[3];
  const float* gfb    = (const float*)d_in[4];
  const float* gbWih  = (const float*)d_in[5];
  const float* gbWhh  = (const float*)d_in[6];
  const float* gbb    = (const float*)d_in[7];
  const float* impW   = (const float*)d_in[8];
  const float* impb   = (const float*)d_in[9];
  const float* fcW    = (const float*)d_in[10];
  const float* fcb    = (const float*)d_in[11];
  const float* decWih = (const float*)d_in[12];
  const float* decWhh = (const float*)d_in[13];
  const float* decb   = (const float*)d_in[14];
  const float* decoW  = (const float*)d_in[15];
  const float* decob  = (const float*)d_in[16];
  const float* dscoW  = (const float*)d_in[17];
  const float* dscob  = (const float*)d_in[18];
  const float* dW0 = (const float*)d_in[19];
  const float* db0 = (const float*)d_in[20];
  const float* dW1 = (const float*)d_in[21];
  const float* db1 = (const float*)d_in[22];
  const float* dW2 = (const float*)d_in[23];
  const float* db2 = (const float*)d_in[24];
  const float* dW3 = (const float*)d_in[25];
  const float* db3 = (const float*)d_in[26];
  const float* dW4 = (const float*)d_in[27];
  const float* db4 = (const float*)d_in[28];

  dim3 grid(BATCH / (16 * 8));  // 256 blocks, 8 waves each, 16 batch rows/wave
  dim3 block(256);
  hipLaunchKernelGGL(crli_kernel, grid, block, 0, stream,
                     values, masks, gfWih, gfWhh, gfb, gbWih, gbWhh, gbb,
                     impW, impb, fcW, fcb, decWih, decWhh, decb, decoW, decob,
                     dscoW, dscob, dW0, db0, dW1, db1, dW2, db2, dW3, db3,
                     dW4, db4, (float*)d_out);
}
